// LocalProcessor_5360119185996
// MI455X (gfx1250) — matrix operand order repaired and checked by reading, not compiled, not run on hardware
//
#include <hip/hip_runtime.h>

typedef __bf16 bf16;
typedef __attribute__((ext_vector_type(8)))  __bf16 v8bf;
typedef __attribute__((ext_vector_type(16))) __bf16 v16bf;
typedef __attribute__((ext_vector_type(8)))  float  v8f;
typedef __attribute__((address_space(3))) char as3char;

// ---------------------------------------------------------------------------
// WMMA fragment helpers (layouts per CDNA5 ISA 7.12.2, wave32)
// A (16x32 bf16): lane L -> row L%16, g=L/16; elems 0..7 -> K=c+g*8+e,
//                 elems 8..15 -> K=c+16+g*8+(e-8)   (two 16B runs)
// B (32x16 bf16): lane L -> col L%16, g=L/16; elems 0..7 -> K=c+g*8+e,
//                 elems 8..15 -> K=c+16+g*8+(e-8) (source K-contiguous per column)
// C/D (16x16 f32): elem r -> row (L<16 ? r : r+8), col L%16
// ---------------------------------------------------------------------------
__device__ __forceinline__ v16bf mk16(v8bf lo, v8bf hi) {
  return __builtin_shufflevector(lo, hi, 0,1,2,3,4,5,6,7,8,9,10,11,12,13,14,15);
}
__device__ __forceinline__ v16bf afrag(const bf16* rowp, int c, int g) {
  v8bf lo = *(const v8bf*)(rowp + c + g * 8);
  v8bf hi = *(const v8bf*)(rowp + c + 16 + g * 8);
  return mk16(lo, hi);
}
__device__ __forceinline__ v16bf bfrag(const bf16* colp, int c, int g) {
  v8bf lo = *(const v8bf*)(colp + c + g * 8);
  v8bf hi = *(const v8bf*)(colp + c + 16 + g * 8);
  return mk16(lo, hi);
}
__device__ __forceinline__ v8f wmma_bf16(v16bf a, v16bf b, v8f c) {
  return __builtin_amdgcn_wmma_f32_16x16x32_bf16(false, a, false, b, (short)0, c,
                                                 false, false);
}

// --- CDNA5 async global->LDS copy (ASYNCcnt-tracked, no VGPR round trip) ----
__device__ __forceinline__ unsigned lds_off(const void* p) {
  return (unsigned)(size_t)(as3char*)p;
}
__device__ __forceinline__ void async_copy_b128(unsigned ldsByteOff,
                                                const void* gptr) {
  asm volatile("global_load_async_to_lds_b128 %0, %1, off"
               :
               : "v"(ldsByteOff), "v"((unsigned long long)(size_t)gptr)
               : "memory");
}
__device__ __forceinline__ void wait_async0() {
  asm volatile("s_wait_asynccnt 0x0" ::: "memory");
}

// ---------------------------------------------------------------------------
// Prep kernels
// ---------------------------------------------------------------------------
// Transposed bf16 weight conversion: dst[n*Kd + k] = src[k*srcLD + n] (0 pad).
__global__ __launch_bounds__(256) void cvtT_k(const float* __restrict__ src,
                                              bf16* __restrict__ dst, int Kv,
                                              int Nv, int Kd, int srcLD,
                                              int total) {
  int i = blockIdx.x * 256 + threadIdx.x;
  if (i >= total) return;
  int n = i / Kd, k = i - n * Kd;
  float v = (k < Kv && n < Nv) ? src[(size_t)k * srcLD + n] : 0.f;
  dst[i] = (bf16)v;
}

// p[b,n,h] = coords . Wc[:,h]  (separable bias; row-constant part cancels in
// softmax, only -p_m survives)
__global__ __launch_bounds__(256) void pcoord_k(const float* __restrict__ coords,
                                                const float* __restrict__ Wc,
                                                float* __restrict__ p, int total) {
  int i = blockIdx.x * 256 + threadIdx.x;
  if (i >= total) return;
  int h = i & 7, bn = i >> 3;
  float acc = 0.f;
#pragma unroll
  for (int c = 0; c < 3; c++) acc += coords[bn * 3 + c] * Wc[c * 8 + h];
  p[i] = acc;
}

// LayerNorm over D=512, one block per row, bf16 out.
__global__ __launch_bounds__(256) void ln_k(const float* __restrict__ x,
                                            const float* __restrict__ w,
                                            const float* __restrict__ b,
                                            bf16* __restrict__ out) {
  const int D = 512;
  int row = blockIdx.x, tid = threadIdx.x;
  const float* xr = x + (size_t)row * D;
  float v0 = xr[tid], v1 = xr[tid + 256];
  float s = v0 + v1, s2 = v0 * v0 + v1 * v1;
#pragma unroll
  for (int off = 16; off; off >>= 1) {
    s += __shfl_xor(s, off, 32);
    s2 += __shfl_xor(s2, off, 32);
  }
  __shared__ float ps[8], ps2[8];
  int wid = tid >> 5, lane = tid & 31;
  if (!lane) { ps[wid] = s; ps2[wid] = s2; }
  __syncthreads();
  float ts = 0.f, ts2 = 0.f;
#pragma unroll
  for (int i = 0; i < 8; i++) { ts += ps[i]; ts2 += ps2[i]; }
  float mean = ts * (1.f / D);
  float var = ts2 * (1.f / D) - mean * mean;
  float inv = rsqrtf(var + 1e-5f);
  bf16* outr = out + (size_t)row * D;
  outr[tid] = (bf16)((v0 - mean) * inv * w[tid] + b[tid]);
  outr[tid + 256] = (bf16)((v1 - mean) * inv * w[tid + 256] + b[tid + 256]);
}

// ---------------------------------------------------------------------------
// bf16 WMMA GEMM: C[M,Nc] = A[M,K] * Bt^T + bias (+res).  Bt is [Nc][K]
// (pre-transposed weights, K-contiguous -> B frags load straight from L2).
// Block 256 thr / 8 waves; block tile 64x128; wave tile 16x64 (4 accs).
// A staged in 64x64 K-stages, double-buffered via async global->LDS: the
// next stage's copy is issued right after the barrier (its target buffer's
// readers are proven done by that barrier) and overlaps this stage's WMMAs.
// EPI 0: bf16 out.  EPI 1: f32 out + residual.  EPI 2: qkv out, V part
// written transposed to vT[b][h*64+d][m] for the attention PV GEMM.
// ---------------------------------------------------------------------------
template <int EPI>
__global__ __launch_bounds__(256) void gemm_k(const bf16* __restrict__ A,
                                              const bf16* __restrict__ Bt,
                                              const float* __restrict__ bias,
                                              const float* __restrict__ res,
                                              float* __restrict__ outF,
                                              bf16* __restrict__ outB,
                                              bf16* __restrict__ vT, int M,
                                              int Nc, int K, int realN) {
  __shared__ __align__(16) bf16 sA[2][64 * 64];
  int tid = threadIdx.x;
  int lane = tid & 31, wid = tid >> 5;
  int g = lane >> 4, j = lane & 15;
  int wm = wid & 3, wn = wid >> 2;
  int bm = blockIdx.y * 64, bn = blockIdx.x * 128;
  // Each thread copies 16 elems of the 64x64 stage: row tid/4, cols (tid&3)*16
  const bf16* aSrc = A + (size_t)(bm + (tid >> 2)) * K + (tid & 3) * 16;
  unsigned rowOff = (tid >> 2) * 128 + (tid & 3) * 32;  // bytes within stage
  unsigned lds0 = lds_off(&sA[0][0]) + rowOff;
  unsigned lds1 = lds_off(&sA[1][0]) + rowOff;
  const bf16* bCol0 = Bt + (size_t)(bn + wn * 64 + j) * K;
  v8f acc0 = {}, acc1 = {}, acc2 = {}, acc3 = {};
  async_copy_b128(lds0, aSrc);
  async_copy_b128(lds0 + 16, aSrc + 8);
  int stages = K >> 6;
  for (int s = 0; s < stages; ++s) {
    int k0 = s << 6;
    wait_async0();
    __syncthreads();
    if (s + 1 < stages) {
      unsigned dst = (s & 1) ? lds0 : lds1;
      async_copy_b128(dst, aSrc + k0 + 64);
      async_copy_b128(dst + 16, aSrc + k0 + 72);
    }
    const bf16* aRow = &sA[s & 1][0] + (wm * 16 + j) * 64;
    v16bf af0 = afrag(aRow, 0, g);
    v16bf af1 = afrag(aRow, 32, g);
    acc0 = wmma_bf16(af0, bfrag(bCol0 + (size_t)0 * K, k0, g), acc0);
    acc0 = wmma_bf16(af1, bfrag(bCol0 + (size_t)0 * K, k0 + 32, g), acc0);
    acc1 = wmma_bf16(af0, bfrag(bCol0 + (size_t)16 * K, k0, g), acc1);
    acc1 = wmma_bf16(af1, bfrag(bCol0 + (size_t)16 * K, k0 + 32, g), acc1);
    acc2 = wmma_bf16(af0, bfrag(bCol0 + (size_t)32 * K, k0, g), acc2);
    acc2 = wmma_bf16(af1, bfrag(bCol0 + (size_t)32 * K, k0 + 32, g), acc2);
    acc3 = wmma_bf16(af0, bfrag(bCol0 + (size_t)48 * K, k0, g), acc3);
    acc3 = wmma_bf16(af1, bfrag(bCol0 + (size_t)48 * K, k0 + 32, g), acc3);
  }
  v8f accs[4] = {acc0, acc1, acc2, acc3};
#pragma unroll
  for (int t = 0; t < 4; t++) {
#pragma unroll
    for (int r = 0; r < 8; r++) {
      int rowIdx = (lane < 16) ? r : r + 8;
      int mrow = bm + wm * 16 + rowIdx;
      int ncol = bn + wn * 64 + t * 16 + j;
      float v = accs[t][r] + ((ncol < realN) ? bias[ncol] : 0.f);
      if (EPI == 1) {
        size_t o = (size_t)mrow * Nc + ncol;
        outF[o] = v + res[o];
      } else if (EPI == 2) {
        if (ncol < 1024) {
          outB[(size_t)mrow * Nc + ncol] = (bf16)v;
        } else {  // V: store transposed for key-contiguous PV fragments
          int hd = ncol - 1024;                  // h*64 + d in [0,512)
          int bb = mrow >> 11, m = mrow & 2047;  // batch, key index
          vT[((size_t)bb * 512 + hd) * 2048 + m] = (bf16)v;
        }
      } else {
        outB[(size_t)mrow * Nc + ncol] = (bf16)v;
      }
    }
  }
}

// ---------------------------------------------------------------------------
// Fused SwiGLU stage 1: ff = silu(A*W1+b1) * (A*W2+b2), bf16 out.
// Same double-buffered async structure; shared A fragments feed both weights.
// ---------------------------------------------------------------------------
__global__ __launch_bounds__(256) void ffn12_k(const bf16* __restrict__ A,
                                               const bf16* __restrict__ W1t,
                                               const bf16* __restrict__ W2t,
                                               const float* __restrict__ b1,
                                               const float* __restrict__ b2,
                                               bf16* __restrict__ ff, int M,
                                               int Nc, int K, int realN) {
  __shared__ __align__(16) bf16 sA[2][64 * 64];
  int tid = threadIdx.x;
  int lane = tid & 31, wid = tid >> 5;
  int g = lane >> 4, j = lane & 15;
  int wm = wid & 3, wn = wid >> 2;
  int bm = blockIdx.y * 64, bn = blockIdx.x * 128;
  const bf16* aSrc = A + (size_t)(bm + (tid >> 2)) * K + (tid & 3) * 16;
  unsigned rowOff = (tid >> 2) * 128 + (tid & 3) * 32;
  unsigned lds0 = lds_off(&sA[0][0]) + rowOff;
  unsigned lds1 = lds_off(&sA[1][0]) + rowOff;
  size_t colBase = (size_t)(bn + wn * 64 + j) * K;
  const bf16* c1 = W1t + colBase;
  const bf16* c2 = W2t + colBase;
  v8f a0 = {}, a1 = {}, a2 = {}, a3 = {};
  v8f d0 = {}, d1 = {}, d2 = {}, d3 = {};
  async_copy_b128(lds0, aSrc);
  async_copy_b128(lds0 + 16, aSrc + 8);
  int stages = K >> 6;
  for (int s = 0; s < stages; ++s) {
    int k0 = s << 6;
    wait_async0();
    __syncthreads();
    if (s + 1 < stages) {
      unsigned dst = (s & 1) ? lds0 : lds1;
      async_copy_b128(dst, aSrc + k0 + 64);
      async_copy_b128(dst + 16, aSrc + k0 + 72);
    }
    const bf16* aRow = &sA[s & 1][0] + (wm * 16 + j) * 64;
    v16bf af0 = afrag(aRow, 0, g);
    v16bf af1 = afrag(aRow, 32, g);
    a0 = wmma_bf16(af0, bfrag(c1 + (size_t)0 * K, k0, g), a0);
    a0 = wmma_bf16(af1, bfrag(c1 + (size_t)0 * K, k0 + 32, g), a0);
    a1 = wmma_bf16(af0, bfrag(c1 + (size_t)16 * K, k0, g), a1);
    a1 = wmma_bf16(af1, bfrag(c1 + (size_t)16 * K, k0 + 32, g), a1);
    a2 = wmma_bf16(af0, bfrag(c1 + (size_t)32 * K, k0, g), a2);
    a2 = wmma_bf16(af1, bfrag(c1 + (size_t)32 * K, k0 + 32, g), a2);
    a3 = wmma_bf16(af0, bfrag(c1 + (size_t)48 * K, k0, g), a3);
    a3 = wmma_bf16(af1, bfrag(c1 + (size_t)48 * K, k0 + 32, g), a3);
    d0 = wmma_bf16(af0, bfrag(c2 + (size_t)0 * K, k0, g), d0);
    d0 = wmma_bf16(af1, bfrag(c2 + (size_t)0 * K, k0 + 32, g), d0);
    d1 = wmma_bf16(af0, bfrag(c2 + (size_t)16 * K, k0, g), d1);
    d1 = wmma_bf16(af1, bfrag(c2 + (size_t)16 * K, k0 + 32, g), d1);
    d2 = wmma_bf16(af0, bfrag(c2 + (size_t)32 * K, k0, g), d2);
    d2 = wmma_bf16(af1, bfrag(c2 + (size_t)32 * K, k0 + 32, g), d2);
    d3 = wmma_bf16(af0, bfrag(c2 + (size_t)48 * K, k0, g), d3);
    d3 = wmma_bf16(af1, bfrag(c2 + (size_t)48 * K, k0 + 32, g), d3);
  }
  v8f ga[4] = {a0, a1, a2, a3};
  v8f gd[4] = {d0, d1, d2, d3};
#pragma unroll
  for (int t = 0; t < 4; t++) {
#pragma unroll
    for (int r = 0; r < 8; r++) {
      int rowIdx = (lane < 16) ? r : r + 8;
      int mrow = bm + wm * 16 + rowIdx;
      int ncol = bn + wn * 64 + t * 16 + j;
      float av = ga[t][r] + ((ncol < realN) ? b1[ncol] : 0.f);
      float cv = gd[t][r] + ((ncol < realN) ? b2[ncol] : 0.f);
      float sig = 1.f / (1.f + __expf(-av));
      ff[(size_t)mrow * Nc + ncol] = (bf16)(av * sig * cv);
    }
  }
}

// ---------------------------------------------------------------------------
// Flash attention: one wave per (b,h, 16 query rows).
// scores = QK^T/sqrt(dh) - p_m  (separable coord bias).  K frags direct from
// qkv (feature-contiguous); V frags direct from pre-transposed vT
// (key-contiguous).  Only P re-layout goes through (per-wave) LDS.
// ---------------------------------------------------------------------------
__global__ __launch_bounds__(128) void attn_k(const bf16* __restrict__ qkv,
                                              const bf16* __restrict__ vT,
                                              const float* __restrict__ pbias,
                                              bf16* __restrict__ obf) {
  const int N = 2048, H = 8, LD = 1536;
  int w = threadIdx.x >> 5;
  int lane = threadIdx.x & 31;
  int g = lane >> 4, j = lane & 15;
  int wg = blockIdx.x * 4 + w;
  int rowTile = wg & 127;
  int bh = wg >> 7;
  int h = bh & 7, b = bh >> 3;

  __shared__ __align__(16) bf16 ldsP[4][16 * 32];
  bf16* myP = ldsP[w];

  const bf16* qbase = qkv + (size_t)b * N * LD + h * 64;
  const bf16* kcol = qbase + 512 + (size_t)j * LD;  // column j of K^T
  const bf16* vbase = vT + ((size_t)b * 512 + h * 64) * 2048;
  const float* pb = pbias + (size_t)b * N * H + h;

  const bf16* qrow = qbase + (size_t)(rowTile * 16 + j) * LD;
  v16bf qa0 = afrag(qrow, 0, g);
  v16bf qa1 = afrag(qrow, 32, g);

  v8f o0 = {}, o1 = {}, o2 = {}, o3 = {};
  float mi[8], li[8];
#pragma unroll
  for (int r = 0; r < 8; r++) { mi[r] = -1e30f; li[r] = 0.f; }
  const float scale = 0.125f;  // 1/sqrt(64)

  for (int m0 = 0; m0 < N; m0 += 32) {
    if (m0 + 32 < N) {  // prefetch next K/V tiles into cache
      __builtin_prefetch(kcol + (size_t)(m0 + 32) * LD, 0, 1);
      __builtin_prefetch(vbase + (size_t)j * 2048 + m0 + 32, 0, 1);
    }
    v8f s0 = {}, s1 = {};
    s0 = wmma_bf16(qa0, bfrag(kcol + (size_t)m0 * LD, 0, g), s0);
    s0 = wmma_bf16(qa1, bfrag(kcol + (size_t)m0 * LD, 32, g), s0);
    s1 = wmma_bf16(qa0, bfrag(kcol + (size_t)(m0 + 16) * LD, 0, g), s1);
    s1 = wmma_bf16(qa1, bfrag(kcol + (size_t)(m0 + 16) * LD, 32, g), s1);
    float pm0 = pb[(size_t)(m0 + j) * H];
    float pm1 = pb[(size_t)(m0 + 16 + j) * H];
#pragma unroll
    for (int r = 0; r < 8; r++) {
      float x0 = s0[r] * scale - pm0;
      float x1 = s1[r] * scale - pm1;
      float mx = fmaxf(x0, x1);
      mx = fmaxf(mx, __shfl_xor(mx, 1, 16));
      mx = fmaxf(mx, __shfl_xor(mx, 2, 16));
      mx = fmaxf(mx, __shfl_xor(mx, 4, 16));
      mx = fmaxf(mx, __shfl_xor(mx, 8, 16));
      float mnew = fmaxf(mi[r], mx);
      float alpha = __expf(mi[r] - mnew);
      mi[r] = mnew;
      float p0 = __expf(x0 - mnew), p1 = __expf(x1 - mnew);
      float rs = p0 + p1;
      rs += __shfl_xor(rs, 1, 16);
      rs += __shfl_xor(rs, 2, 16);
      rs += __shfl_xor(rs, 4, 16);
      rs += __shfl_xor(rs, 8, 16);
      li[r] = li[r] * alpha + rs;
      o0[r] *= alpha; o1[r] *= alpha; o2[r] *= alpha; o3[r] *= alpha;
      int rowIdx = (lane < 16) ? r : r + 8;
      myP[rowIdx * 32 + j] = (bf16)p0;
      myP[rowIdx * 32 + 16 + j] = (bf16)p1;
    }
    v16bf pa = afrag(myP + j * 32, 0, g);  // LDS in-order within a wave
    o0 = wmma_bf16(pa, bfrag(vbase + (size_t)(0 + j) * 2048, m0, g), o0);
    o1 = wmma_bf16(pa, bfrag(vbase + (size_t)(16 + j) * 2048, m0, g), o1);
    o2 = wmma_bf16(pa, bfrag(vbase + (size_t)(32 + j) * 2048, m0, g), o2);
    o3 = wmma_bf16(pa, bfrag(vbase + (size_t)(48 + j) * 2048, m0, g), o3);
  }
#pragma unroll
  for (int r = 0; r < 8; r++) {
    int rowIdx = (lane < 16) ? r : r + 8;
    size_t base = ((size_t)b * N + rowTile * 16 + rowIdx) * 512 + h * 64;
    float inv = 1.f / li[r];
    obf[base + 0 + j] = (bf16)(o0[r] * inv);
    obf[base + 16 + j] = (bf16)(o1[r] * inv);
    obf[base + 32 + j] = (bf16)(o2[r] * inv);
    obf[base + 48 + j] = (bf16)(o3[r] * inv);
  }
}

// ---------------------------------------------------------------------------
extern "C" void kernel_launch(void* const* d_in, const int* in_sizes, int n_in,
                              void* d_out, int out_size, void* d_ws,
                              size_t ws_size, hipStream_t stream) {
  (void)in_sizes; (void)n_in; (void)out_size; (void)ws_size;
  const float* x = (const float*)d_in[0];
  const float* coords = (const float*)d_in[1];
  const float* ln1w = (const float*)d_in[2];
  const float* ln1b = (const float*)d_in[3];
  const float* ln2w = (const float*)d_in[4];
  const float* ln2b = (const float*)d_in[5];
  const float* Wqkv = (const float*)d_in[6];
  const float* bqkv = (const float*)d_in[7];
  const float* Wo = (const float*)d_in[8];
  const float* bo = (const float*)d_in[9];
  const float* Wc = (const float*)d_in[10];
  const float* W1 = (const float*)d_in[12];
  const float* b1 = (const float*)d_in[13];
  const float* W2 = (const float*)d_in[14];
  const float* b2 = (const float*)d_in[15];
  const float* W3 = (const float*)d_in[16];
  const float* b3 = (const float*)d_in[17];
  float* out = (float*)d_out;

  const int M = 8192;  // B*N
  const int D = 512, QKVN = 1536, HID = 1365, HP = 1408;

  char* ws = (char*)d_ws;
  auto alloc = [&](size_t bytes) {
    char* p = ws;
    ws += (bytes + 255) & ~(size_t)255;
    return p;
  };
  float* pW = (float*)alloc((size_t)M * 8 * 4);
  bf16* hbf = (bf16*)alloc((size_t)M * D * 2);
  bf16* qkvbf = (bf16*)alloc((size_t)M * QKVN * 2);
  bf16* vTw = (bf16*)alloc((size_t)4 * 512 * 2048 * 2);
  bf16* obf = (bf16*)alloc((size_t)M * D * 2);
  float* x1 = (float*)alloc((size_t)M * D * 4);
  bf16* h2bf = (bf16*)alloc((size_t)M * D * 2);
  bf16* ffbf = (bf16*)alloc((size_t)M * HP * 2);
  bf16* WqkvT = (bf16*)alloc((size_t)QKVN * D * 2);
  bf16* WoT = (bf16*)alloc((size_t)D * D * 2);
  bf16* W1T = (bf16*)alloc((size_t)HP * D * 2);
  bf16* W2T = (bf16*)alloc((size_t)HP * D * 2);
  bf16* W3T = (bf16*)alloc((size_t)D * HP * 2);

  auto blks = [](int total) { return (total + 255) / 256; };
  // Transposed bf16 weight conversion (zero-padded HID -> HP)
  cvtT_k<<<blks(QKVN * D), 256, 0, stream>>>(Wqkv, WqkvT, D, QKVN, D, QKVN, QKVN * D);
  cvtT_k<<<blks(D * D), 256, 0, stream>>>(Wo, WoT, D, D, D, D, D * D);
  cvtT_k<<<blks(HP * D), 256, 0, stream>>>(W1, W1T, D, HID, D, HID, HP * D);
  cvtT_k<<<blks(HP * D), 256, 0, stream>>>(W2, W2T, D, HID, D, HID, HP * D);
  cvtT_k<<<blks(D * HP), 256, 0, stream>>>(W3, W3T, HID, D, HP, D, D * HP);
  // Separable attention bias precompute
  pcoord_k<<<blks(M * 8), 256, 0, stream>>>(coords, Wc, pW, M * 8);
  // LN1 -> bf16
  ln_k<<<M, 256, 0, stream>>>(x, ln1w, ln1b, hbf);
  // QKV projection; V part stored transposed for attention
  gemm_k<2><<<dim3(QKVN / 128, M / 64), 256, 0, stream>>>(
      hbf, WqkvT, bqkv, nullptr, nullptr, qkvbf, vTw, M, QKVN, D, QKVN);
  // Flash attention (4096 waves, 4/block)
  attn_k<<<1024, 128, 0, stream>>>(qkvbf, vTw, pW, obf);
  // Output projection + residual 1
  gemm_k<1><<<dim3(D / 128, M / 64), 256, 0, stream>>>(
      obf, WoT, bo, x, x1, nullptr, nullptr, M, D, D, D);
  // LN2 -> bf16
  ln_k<<<M, 256, 0, stream>>>(x1, ln2w, ln2b, h2bf);
  // SwiGLU stage 1 (fused dual GEMM)
  ffn12_k<<<dim3(HP / 128, M / 64), 256, 0, stream>>>(
      h2bf, W1T, W2T, b1, b2, ffbf, M, HP, D, HID);
  // FFN down-projection + residual 2 -> f32 output
  gemm_k<1><<<dim3(D / 128, M / 64), 256, 0, stream>>>(
      ffbf, W3T, b3, x1, out, nullptr, nullptr, M, D, HP, D);
}
